// VectorQuantizer_49615462203803
// MI455X (gfx1250) — matrix operand order repaired and checked by reading, not compiled, not run on hardware
//
#include <hip/hip_runtime.h>
#include <hip/hip_bf16.h>

typedef __attribute__((ext_vector_type(16))) _Float16 v16h;
typedef __attribute__((ext_vector_type(8)))  _Float16 v8h;
typedef __attribute__((ext_vector_type(8)))  float    v8f;
typedef int vi4 __attribute__((vector_size(16)));   // b128 payload type for async builtins

#define KCODES 1024
#define DDIM   256
#define BB     32
#define HH     32
#define WW     32
#define NPTS   (BB * HH * WW)   // 32768
#define NCHUNK (KCODES / 16)    // 64

#define AS1 __attribute__((address_space(1)))
#define AS3 __attribute__((address_space(3)))

// CDNA5 async global->LDS path (ASYNCcnt-tracked), with manual fallback.
#if defined(__gfx1250__) && __has_builtin(__builtin_amdgcn_global_load_async_to_lds_b128)
#define VQ_ASYNC 1
#else
#define VQ_ASYNC 0
#endif

#if VQ_ASYNC
#define VQ_GLD_ASYNC_B128(gp, lp, off) \
    __builtin_amdgcn_global_load_async_to_lds_b128( \
        (AS1 vi4*)(gp), (AS3 vi4*)(lp), (off), 0)

__device__ __forceinline__ void vq_wait_async0() {
#if __has_builtin(__builtin_amdgcn_s_wait_asynccnt)
    __builtin_amdgcn_s_wait_asynccnt(0);
#else
    asm volatile("s_wait_asynccnt 0x0" ::: "memory");
#endif
}
#endif

// ---------------------------------------------------------------------------
// Kernel 1: codebook prep.  cb16[k][d] = (f16)(-2*c[k][d]);  csq[k] = ||c_k||^2
// ---------------------------------------------------------------------------
__global__ void vq_prep_codebook(const float* __restrict__ cb,
                                 _Float16* __restrict__ cb16,
                                 float* __restrict__ csq) {
    const int k = blockIdx.x;
    const int d = threadIdx.x;
    float v = cb[k * DDIM + d];
    cb16[k * DDIM + d] = (_Float16)(-2.0f * v);

    float s = v * v;
    #pragma unroll
    for (int m = 16; m >= 1; m >>= 1)
        s += __shfl_xor(s, m, 32);

    __shared__ float wsum[8];
    if ((threadIdx.x & 31) == 0) wsum[threadIdx.x >> 5] = s;
    __syncthreads();
    if (threadIdx.x == 0) {
        float t = 0.0f;
        #pragma unroll
        for (int i = 0; i < 8; ++i) t += wsum[i];
        csq[k] = t;
    }
}

// ---------------------------------------------------------------------------
// Kernel 2: x [B,D,H,W] fp32  ->  xf16 [N, D] fp16 (NHWC flatten), LDS transpose.
// ---------------------------------------------------------------------------
__global__ void vq_prep_x(const float* __restrict__ x,
                          _Float16* __restrict__ xf16) {
    const int bh = blockIdx.x;          // 0 .. B*H-1
    const int b  = bh >> 5;
    const int h  = bh & 31;
    const int tx = threadIdx.x & 31;
    const int ty = threadIdx.x >> 5;

    __shared__ _Float16 tile[32 * 33];

    const long HW    = (long)HH * WW;
    const long xbase = ((long)b * DDIM) * HW + (long)h * WW;   // + d*HW + w
    const long nbase = ((long)b * HH + h) * WW;                // + w

    for (int p = 0; p < DDIM / 32; ++p) {
        __syncthreads();
        #pragma unroll
        for (int i = 0; i < 4; ++i) {
            const int dl = ty + 8 * i;
            const int d  = p * 32 + dl;
            tile[dl * 33 + tx] = (_Float16)x[xbase + (long)d * HW + tx];
        }
        __syncthreads();
        #pragma unroll
        for (int i = 0; i < 4; ++i) {
            const int wl = ty + 8 * i;
            xf16[(nbase + wl) * DDIM + p * 32 + tx] = tile[tx * 33 + wl];
        }
    }
}

// ---------------------------------------------------------------------------
// Kernel 3: fused GEMM + argmin.
// ---------------------------------------------------------------------------
__device__ __forceinline__ v16h vq_bfrag(const v8h* bp, int idx) {
    v8h lo = bp[idx];
    v8h hi = bp[idx + 2];
    return __builtin_shufflevector(lo, hi,
            0, 1, 2, 3, 4, 5, 6, 7, 8, 9, 10, 11, 12, 13, 14, 15);
}

__global__ __launch_bounds__(256) void vq_argmin(
        const _Float16* __restrict__ xf16,
        const _Float16* __restrict__ cb16,
        const float* __restrict__ csq,
        int* __restrict__ out) {

    __shared__ float4 lbuf[2][512];     // 2 x 8KB codebook chunk buffers
    __shared__ float  scsq[KCODES];     // 4KB  ||c_k||^2 table

    const int tid  = threadIdx.x;
    const int wv   = tid >> 5;
    const int lane = tid & 31;
    const int l15  = lane & 15;
    const int sel  = lane >> 4;         // 0: low half-wave, 1: high

    const int row_base = blockIdx.x * 128 + wv * 16;
    const int row      = row_base + l15;    // A: lanes m and m+16 hold row m

    // ---- Preload all A fragments for this wave's 16 rows (K = 0..255) ----
    // ISA 16-bit A 16x32 layout: lane<16 -> K {0..7,16..23}; lane>=16 -> {8..15,24..31}
    v16h a[8];
    const v8h* rp = (const v8h*)(xf16 + (long)row * DDIM);
    #pragma unroll
    for (int ks = 0; ks < 8; ++ks) {
        v8h lo = rp[ks * 4 + sel];          // d = ks*32 + sel*8
        v8h hi = rp[ks * 4 + 2 + sel];      // d = ks*32 + 16 + sel*8
        a[ks] = __builtin_shufflevector(lo, hi,
                0, 1, 2, 3, 4, 5, 6, 7, 8, 9, 10, 11, 12, 13, 14, 15);
    }

    // ---- Stage csq + chunk 0 ----
    const float4* cbv = (const float4*)cb16;    // chunk c = cbv[c*512 .. +512)
    #pragma unroll
    for (int i = 0; i < KCODES / 256; ++i)
        scsq[tid + 256 * i] = csq[tid + 256 * i];
#if VQ_ASYNC
    VQ_GLD_ASYNC_B128(cbv + tid * 2, &lbuf[0][tid * 2], 0);
    VQ_GLD_ASYNC_B128(cbv + tid * 2, &lbuf[0][tid * 2], 16);
    vq_wait_async0();
#else
    lbuf[0][tid * 2 + 0] = cbv[tid * 2 + 0];
    lbuf[0][tid * 2 + 1] = cbv[tid * 2 + 1];
#endif
    __syncthreads();

    float bv[8];
    int   bi[8];
    #pragma unroll
    for (int j = 0; j < 8; ++j) { bv[j] = 3.4e38f; bi[j] = 0; }

    for (int c = 0; c < NCHUNK; ++c) {
        // ---- Kick off staging of the next chunk (overlaps with compute) ----
#if VQ_ASYNC
        if (c + 1 < NCHUNK) {
            VQ_GLD_ASYNC_B128(cbv + (c + 1) * 512 + tid * 2,
                              &lbuf[(c + 1) & 1][tid * 2], 0);
            VQ_GLD_ASYNC_B128(cbv + (c + 1) * 512 + tid * 2,
                              &lbuf[(c + 1) & 1][tid * 2], 16);
        }
#else
        float4 p0, p1;
        if (c + 1 < NCHUNK) {
            p0 = cbv[(c + 1) * 512 + tid * 2 + 0];
            p1 = cbv[(c + 1) * 512 + tid * 2 + 1];
        }
#endif

        const v8h* bp   = (const v8h*)(&lbuf[c & 1][0]);
        const int  code = c * 16 + l15;             // this lane's column
        const float cv  = scsq[code];

        // Two independent accumulator chains, 4 live B fragments (loads run
        // >=2 WMMAs ahead of their consumers).
        v8f acc0 = { cv, cv, cv, cv, cv, cv, cv, cv };  // seeded with ||c||^2
        v8f acc1 = { 0.f, 0.f, 0.f, 0.f, 0.f, 0.f, 0.f, 0.f };
        const int bbase = l15 * 32 + sel;

        v16h b0 = vq_bfrag(bp, bbase + 0 * 4);
        v16h b1 = vq_bfrag(bp, bbase + 1 * 4);
        v16h b2 = vq_bfrag(bp, bbase + 2 * 4);
        v16h b3 = vq_bfrag(bp, bbase + 3 * 4);
        acc0 = __builtin_amdgcn_wmma_f32_16x16x32_f16(false, a[0], false, b0, (short)0, acc0, false, false);
        acc1 = __builtin_amdgcn_wmma_f32_16x16x32_f16(false, a[1], false, b1, (short)0, acc1, false, false);
        b0 = vq_bfrag(bp, bbase + 4 * 4);
        b1 = vq_bfrag(bp, bbase + 5 * 4);
        acc0 = __builtin_amdgcn_wmma_f32_16x16x32_f16(false, a[2], false, b2, (short)0, acc0, false, false);
        acc1 = __builtin_amdgcn_wmma_f32_16x16x32_f16(false, a[3], false, b3, (short)0, acc1, false, false);
        b2 = vq_bfrag(bp, bbase + 6 * 4);
        b3 = vq_bfrag(bp, bbase + 7 * 4);
        acc0 = __builtin_amdgcn_wmma_f32_16x16x32_f16(false, a[4], false, b0, (short)0, acc0, false, false);
        acc1 = __builtin_amdgcn_wmma_f32_16x16x32_f16(false, a[5], false, b1, (short)0, acc1, false, false);
        acc0 = __builtin_amdgcn_wmma_f32_16x16x32_f16(false, a[6], false, b2, (short)0, acc0, false, false);
        acc1 = __builtin_amdgcn_wmma_f32_16x16x32_f16(false, a[7], false, b3, (short)0, acc1, false, false);

        // Branch-free running argmin (strict < keeps the earliest index).
        #pragma unroll
        for (int j = 0; j < 8; ++j) {
            const float d2 = acc0[j] + acc1[j];
            const bool  lt = d2 < bv[j];
            bv[j] = lt ? d2   : bv[j];
            bi[j] = lt ? code : bi[j];
        }

        // Commit next chunk, then sync the iteration.
#if VQ_ASYNC
        if (c + 1 < NCHUNK) vq_wait_async0();
#else
        if (c + 1 < NCHUNK) {
            float4* dst = &lbuf[(c + 1) & 1][0];
            dst[tid * 2 + 0] = p0;
            dst[tid * 2 + 1] = p1;
        }
#endif
        __syncthreads();
    }

    // ---- Argmin across the 16 lanes of each half-wave (cols mod 16) ----
    #pragma unroll
    for (int j = 0; j < 8; ++j) {
        #pragma unroll
        for (int m = 8; m >= 1; m >>= 1) {
            const float ov = __shfl_xor(bv[j], m, 32);
            const int   oi = __shfl_xor(bi[j], m, 32);
            const bool take = (ov < bv[j]) || (ov == bv[j] && oi < bi[j]);
            bv[j] = take ? ov : bv[j];
            bi[j] = take ? oi : bi[j];
        }
    }

    // C/D layout: VGPR j = row j (lanes 0-15) / row j+8 (lanes 16-31).
    if (l15 == 0) {
        const int rb = row_base + sel * 8;
        #pragma unroll
        for (int j = 0; j < 8; ++j)
            out[rb + j] = bi[j];
    }
}

// ---------------------------------------------------------------------------
extern "C" void kernel_launch(void* const* d_in, const int* in_sizes, int n_in,
                              void* d_out, int out_size, void* d_ws, size_t ws_size,
                              hipStream_t stream) {
    (void)in_sizes; (void)n_in; (void)out_size; (void)ws_size;

    const float* x  = (const float*)d_in[0];   // [B, D, H, W] fp32
    const float* cb = (const float*)d_in[1];   // [K, D] fp32

    char* ws = (char*)d_ws;
    _Float16* xf16 = (_Float16*)ws;                                   // 16 MB
    _Float16* cb16 = (_Float16*)(ws + (size_t)NPTS * DDIM * 2);       // 512 KB
    float*    csq  = (float*)(ws + (size_t)NPTS * DDIM * 2
                                 + (size_t)KCODES * DDIM * 2);        // 4 KB
    int* out = (int*)d_out;

    vq_prep_codebook<<<KCODES, 256, 0, stream>>>(cb, cb16, csq);
    vq_prep_x<<<BB * HH, 256, 0, stream>>>(x, xf16);
    vq_argmin<<<NPTS / 128, 256, 0, stream>>>(xf16, cb16, csq, out);
}
